// GNN_38474317037945
// MI455X (gfx1250) — matrix operand order repaired and checked by reading, not compiled, not run on hardware
//
#include <hip/hip_runtime.h>

typedef __bf16 bf16;
typedef __attribute__((ext_vector_type(16))) __bf16 bf16x16;
typedef __attribute__((ext_vector_type(8)))  float  v8f;

#define N_NODES 8192
#define KNN     16
#define NGRAPH  32
#define EPSB    1e-5f
#define INF_F   3.0e38f
#define MAXC    768     // max column span (two adjacent graphs ~512 avg nodes)

union FragB16 { bf16x16 v; uint4 q[2]; };

// CDNA5 async global->LDS copy (ASYNCcnt-tracked), 16 bytes per lane.
// vdst = 32-bit LDS byte offset, vaddr = 64-bit global address.
#define ASYNC_LD128(ldsOff, gptr)                                        \
  asm volatile("global_load_async_to_lds_b128 %0, %1, off"               \
               :: "v"(ldsOff), "v"(gptr) : "memory")
#define WAIT_ASYNC() asm volatile("s_wait_asynccnt 0x0" ::: "memory")

__device__ __forceinline__ unsigned lds_off(const void* p) {
  // flat shared address = {shared aperture, 32-bit LDS offset}
  return (unsigned)(size_t)p;
}

// ---------------------------------------------------------------- utilities
__global__ void zero_f32(float* p, int n) {
  int i = blockIdx.x * blockDim.x + threadIdx.x;
  if (i < n) p[i] = 0.f;
}

__global__ void f32_to_bf16(const float* __restrict__ x, bf16* __restrict__ y, int n) {
  int i = blockIdx.x * blockDim.x + threadIdx.x;
  if (i < n) y[i] = (bf16)x[i];
}

// W:[K][N] fp32 -> Wt:[N][K] bf16 (B operands are consumed transposed)
__global__ void transpose_to_bf16(const float* __restrict__ W, bf16* __restrict__ Wt,
                                  int K, int N) {
  int i = blockIdx.x * blockDim.x + threadIdx.x;
  if (i >= K * N) return;
  int k = i / N, n = i % N;
  Wt[(size_t)n * K + k] = (bf16)W[i];
}

// factorized edge-message weights, transposed: Ut,Vt:[co][ci]
__global__ void prep_conv_w(const float* __restrict__ W, bf16* __restrict__ Ut,
                            bf16* __restrict__ Vt, int ci, int co) {
  int i = blockIdx.x * blockDim.x + threadIdx.x;
  if (i >= ci * co) return;
  int r = i / co, c = i % co;
  float wt = W[(size_t)r * co + c];
  float wb = W[(size_t)(ci + r) * co + c];
  Ut[(size_t)c * ci + r] = (bf16)(wt - wb);
  Vt[(size_t)c * ci + r] = (bf16)wb;
}

// per-node graph range [gs,ge) by binary search over sorted batch vector
__global__ void ranges_kernel(const int* __restrict__ batch, int* __restrict__ gs,
                              int* __restrict__ ge, int n) {
  int i = blockIdx.x * blockDim.x + threadIdx.x;
  if (i >= n) return;
  int b = batch[i];
  int lo = 0, hi = n;
  while (lo < hi) { int m = (lo + hi) >> 1; if (batch[m] < b) lo = m + 1; else hi = m; }
  gs[i] = lo;
  lo = 0; hi = n;
  while (lo < hi) { int m = (lo + hi) >> 1; if (batch[m] <= b) lo = m + 1; else hi = m; }
  ge[i] = lo;
}

// squared row norms of bf16 feature matrix
__global__ void rownorm2(const bf16* __restrict__ h, float* __restrict__ d2,
                         int n, int c) {
  int i = blockIdx.x * blockDim.x + threadIdx.x;
  if (i >= n) return;
  const bf16* row = h + (size_t)i * c;
  float s = 0.f;
  for (int j = 0; j < c; ++j) { float v = (float)row[j]; s += v * v; }
  d2[i] = s;
}

// column sum / sumsq of fp32 [rows][C] into stats[0..C) / stats[C..2C)
__global__ void colstats(const float* __restrict__ X, float* __restrict__ stats,
                         int rows, int C) {
  int c = blockIdx.x * blockDim.x + threadIdx.x;
  if (c >= C) return;
  int per = rows / gridDim.y;
  int r0 = blockIdx.y * per;
  float s = 0.f, ss = 0.f;
  for (int r = r0; r < r0 + per; ++r) {
    float v = X[(size_t)r * C + c];
    s += v; ss += v * v;
  }
  atomicAdd(&stats[c], s);
  atomicAdd(&stats[C + c], ss);
}

// y = relu((x-mean)*rsqrt(var+eps)*g + b) -> bf16
__global__ void bn_relu_bf16(const float* __restrict__ X, const float* __restrict__ stats,
                             const float* __restrict__ g, const float* __restrict__ b,
                             bf16* __restrict__ Y, int rows, int C) {
  int i = blockIdx.x * blockDim.x + threadIdx.x;
  if (i >= rows * C) return;
  int c = i % C;
  float inv = 1.f / (float)rows;
  float m = stats[c] * inv;
  float var = stats[C + c] * inv - m * m;
  float v = (X[i] - m) * rsqrtf(var + EPSB) * g[c] + b[c];
  Y[i] = (bf16)fmaxf(v, 0.f);
}

// BN statistics over the N*K edge-message rows: Z[n,k,c] = P[n,c] + Q[idx[n,k],c]
__global__ void msg_stats(const float* __restrict__ P, const float* __restrict__ Q,
                          const int* __restrict__ knn, float* __restrict__ stats, int C) {
  int c = blockIdx.x * blockDim.x + threadIdx.x;
  if (c >= C) return;
  int total = N_NODES * KNN;
  int per = total / gridDim.y;
  int e0 = blockIdx.y * per;
  float s = 0.f, ss = 0.f;
  for (int e = e0; e < e0 + per; ++e) {
    int n = e >> 4;
    int j = knn[e];
    float z = P[(size_t)n * C + c] + Q[(size_t)j * C + c];
    s += z; ss += z * z;
  }
  atomicAdd(&stats[c], s);
  atomicAdd(&stats[C + c], ss);
}

// out[n,c] = max_k relu( bn(P[n,c]+Q[idx[n,k],c]) )
__global__ void msg_max_relu(const float* __restrict__ P, const float* __restrict__ Q,
                             const int* __restrict__ knn, const float* __restrict__ stats,
                             const float* __restrict__ g, const float* __restrict__ b,
                             float* __restrict__ out, int C) {
  int i = blockIdx.x * blockDim.x + threadIdx.x;
  if (i >= N_NODES * C) return;
  int n = i / C, c = i % C;
  const float inv = 1.f / (float)(N_NODES * KNN);
  float m = stats[c] * inv;
  float var = stats[C + c] * inv - m * m;
  float rs = rsqrtf(var + EPSB) * g[c];
  float bb = b[c];
  float p = P[i];
  float best = 0.f;
  #pragma unroll
  for (int k = 0; k < KNN; ++k) {
    int j = knn[n * KNN + k];
    float z = (p + Q[(size_t)j * C + c] - m) * rs + bb;
    best = fmaxf(best, fmaxf(z, 0.f));
  }
  out[i] = best;
}

// ---------------------------------------------------------------- dense GEMM
// C[M,N] = A[M,K](bf16,row) @ Bt[N,K](bf16,row, i.e. B transposed) + bias.
// 64x64 block tile, 4 waves, double-buffered async global->LDS staging.
__global__ __launch_bounds__(128)
void gemm_bf16_wmma(const bf16* __restrict__ A, const bf16* __restrict__ Bt,
                    const float* __restrict__ bias, float* __restrict__ C,
                    int M, int N, int K) {
  __shared__ __align__(16) bf16 sA[2][64][32];
  __shared__ __align__(16) bf16 sB[2][64][32];

  const int tid = threadIdx.x;
  const int wv = tid >> 5, lane = tid & 31;
  const int l15 = lane & 15, hf = lane >> 4;
  const int bm = blockIdx.x * 64, bn = blockIdx.y * 64;

  // staging assignment: thread -> (row, row+32) x 16B chunk
  const int row = tid >> 2;
  const int ko  = (tid & 3) * 8;

  const bf16* gA0 = A  + (size_t)(bm + row) * K + ko;
  const bf16* gA1 = A  + (size_t)(bm + 32 + row) * K + ko;
  const bf16* gB0 = Bt + (size_t)(bn + row) * K + ko;
  const bf16* gB1 = Bt + (size_t)(bn + 32 + row) * K + ko;

  v8f acc[4] = {};

  // prologue: stage tile 0 into buffer 0
  ASYNC_LD128(lds_off(&sA[0][row][ko]),      gA0);
  ASYNC_LD128(lds_off(&sA[0][32 + row][ko]), gA1);
  ASYNC_LD128(lds_off(&sB[0][row][ko]),      gB0);
  ASYNC_LD128(lds_off(&sB[0][32 + row][ko]), gB1);
  WAIT_ASYNC();
  __syncthreads();

  int cur = 0;
  for (int k0 = 0; k0 < K; k0 += 32) {
    int nxt = cur ^ 1;
    if (k0 + 32 < K) {            // overlap next tile copy with current WMMAs
      ASYNC_LD128(lds_off(&sA[nxt][row][ko]),      gA0 + k0 + 32);
      ASYNC_LD128(lds_off(&sA[nxt][32 + row][ko]), gA1 + k0 + 32);
      ASYNC_LD128(lds_off(&sB[nxt][row][ko]),      gB0 + k0 + 32);
      ASYNC_LD128(lds_off(&sB[nxt][32 + row][ko]), gB1 + k0 + 32);
    }

    FragB16 a, b;
    a.q[0] = *(const uint4*)&sA[cur][(wv << 4) + l15][hf * 8];
    a.q[1] = *(const uint4*)&sA[cur][(wv << 4) + l15][16 + hf * 8];
    #pragma unroll
    for (int f = 0; f < 4; ++f) {
      b.q[0] = *(const uint4*)&sB[cur][f * 16 + l15][hf * 8];
      b.q[1] = *(const uint4*)&sB[cur][f * 16 + l15][16 + hf * 8];
      acc[f] = __builtin_amdgcn_wmma_f32_16x16x32_bf16(
          false, a.v, false, b.v, (short)0, acc[f], false, false);
    }

    WAIT_ASYNC();        // own async issues retired
    __syncthreads();     // whole tile staged + all waves done reading 'cur'
    cur = nxt;
  }

  #pragma unroll
  for (int f = 0; f < 4; ++f) {
    int col = bn + f * 16 + l15;
    float bv = bias ? bias[col] : 0.f;
    int row0 = bm + (wv << 4) + hf * 8;
    #pragma unroll
    for (int r = 0; r < 8; ++r)
      C[(size_t)(row0 + r) * N + col] = acc[f][r] + bv;
  }
}

// ---------------------------------------------------------------- fused kNN
// One wave per 16 rows.  WMMA gram tile vs the rows' graph column range only,
// distances staged in LDS, then 16-way serial min-selection of K neighbors.
__global__ __launch_bounds__(32)
void knn_wmma(const bf16* __restrict__ H, const float* __restrict__ d2,
              const int* __restrict__ batch, const int* __restrict__ gs,
              const int* __restrict__ ge, int* __restrict__ knn, int D) {
  __shared__ __align__(16) float dist[16][MAXC];     // 48 KB
  const int lane = threadIdx.x;
  const int l15 = lane & 15, hf = lane >> 4;
  const int rb = blockIdx.x * 16;

  int cs = 0x7fffffff, ce = 0;
  for (int r = 0; r < 16; ++r) {
    cs = min(cs, gs[rb + r]);
    ce = max(ce, ge[rb + r]);
  }
  if (ce - cs > MAXC) ce = cs + MAXC;

  const int mr = hf * 8;                // C-fragment row base for this lane
  int bi[8]; float di[8];
  #pragma unroll
  for (int r = 0; r < 8; ++r) {
    bi[r] = batch[rb + mr + r];
    di[r] = d2[rb + mr + r];
  }

  const bf16* arow = H + (size_t)(rb + l15) * D;

  for (int c0 = cs; c0 < ce; c0 += 16) {
    int j = c0 + l15;
    bool jv = (j < ce);
    const bf16* brow = H + (size_t)(jv ? j : cs) * D; // B col n = neighbor's feature row
    v8f acc = {};
    for (int k = 0; k < D; k += 32) {
      FragB16 a, b;
      a.q[0] = *(const uint4*)(arow + k + hf * 8);
      a.q[1] = *(const uint4*)(arow + k + 16 + hf * 8);
      b.q[0] = *(const uint4*)(brow + k + hf * 8);
      b.q[1] = *(const uint4*)(brow + k + 16 + hf * 8);
      acc = __builtin_amdgcn_wmma_f32_16x16x32_bf16(
          false, a.v, false, b.v, (short)0, acc, false, false);
    }
    if (jv) {
      int jb = batch[j];
      float jd = d2[j];
      #pragma unroll
      for (int r = 0; r < 8; ++r) {
        float dd = di[r] + jd - 2.f * acc[r];
        if (jb != bi[r]) dd = INF_F;
        dist[mr + r][j - cs] = dd;
      }
    }
  }
  __syncthreads();

  if (lane < 16) {
    int len = ce - cs;
    float* row = dist[lane];
    int base = (rb + lane) * KNN;
    for (int t = 0; t < KNN; ++t) {
      float best = INF_F; int bj = 0;
      for (int q = 0; q < len; ++q) {
        float v = row[q];
        if (v < best) { best = v; bj = q; }
      }
      knn[base + t] = cs + bj;
      row[bj] = INF_F;
    }
  }
}

// ---------------------------------------------------------------- pooling
__global__ void pool_accum(const float* __restrict__ F, const int* __restrict__ batch,
                           float* __restrict__ out, float* __restrict__ cnt) {
  int i = blockIdx.x * blockDim.x + threadIdx.x;      // N_NODES*256 threads
  int n = i >> 8, c = i & 255;
  int b = batch[n];
  atomicAdd(&out[b * 256 + c], F[i]);
  if (c == 0) atomicAdd(&cnt[b], 1.f);
}

__global__ void pool_div(float* __restrict__ out, const float* __restrict__ cnt) {
  int b = blockIdx.x, c = threadIdx.x;
  out[b * 256 + c] /= fmaxf(cnt[b], 1.f);
}

// ---------------------------------------------------------------- driver
extern "C" void kernel_launch(void* const* d_in, const int* in_sizes, int n_in,
                              void* d_out, int out_size, void* d_ws, size_t ws_size,
                              hipStream_t stream) {
  (void)in_sizes; (void)n_in; (void)out_size; (void)ws_size;
  const float* x     = (const float*)d_in[0];
  const int*   batch = (const int*)d_in[4];
  const float* nW1  = (const float*)d_in[5];
  const float* nb1  = (const float*)d_in[6];
  const float* ng1  = (const float*)d_in[7];
  const float* nbe1 = (const float*)d_in[8];
  const float* nW2  = (const float*)d_in[9];
  const float* nb2  = (const float*)d_in[10];
  // edge MLP (d_in[11..16]) has no effect on the output: skipped
  const float* cW[3]  = {(const float*)d_in[17], (const float*)d_in[23], (const float*)d_in[29]};
  const float* cb[3]  = {(const float*)d_in[18], (const float*)d_in[24], (const float*)d_in[30]};
  const float* cg[3]  = {(const float*)d_in[19], (const float*)d_in[25], (const float*)d_in[31]};
  const float* cbe[3] = {(const float*)d_in[20], (const float*)d_in[26], (const float*)d_in[32]};
  const float* cgn[3] = {(const float*)d_in[21], (const float*)d_in[27], (const float*)d_in[33]};
  const float* cbn[3] = {(const float*)d_in[22], (const float*)d_in[28], (const float*)d_in[34]};
  const float* mW1  = (const float*)d_in[35];
  const float* mb1  = (const float*)d_in[36];
  const float* mg1  = (const float*)d_in[37];
  const float* mbe1 = (const float*)d_in[38];
  const float* mW2  = (const float*)d_in[39];
  const float* mb2  = (const float*)d_in[40];

  char* w = (char*)d_ws;
  auto carve = [&](size_t bytes) { char* p = w; w += (bytes + 255) & ~(size_t)255; return p; };
  float* bufA  = (float*)carve((size_t)N_NODES * 512 * 4);
  float* bufB  = (float*)carve((size_t)N_NODES * 512 * 4);
  float* bufP  = (float*)carve((size_t)N_NODES * 512 * 4);
  float* bufQ  = (float*)carve((size_t)N_NODES * 512 * 4);
  bf16*  hbf   = (bf16*) carve((size_t)N_NODES * 512 * 2);
  bf16*  WU    = (bf16*) carve((size_t)512 * 512 * 2);
  bf16*  WV    = (bf16*) carve((size_t)512 * 512 * 2);
  int*   idx   = (int*)  carve((size_t)N_NODES * KNN * 4);
  float* d2    = (float*)carve((size_t)N_NODES * 4);
  int*   gs    = (int*)  carve((size_t)N_NODES * 4);
  int*   ge    = (int*)  carve((size_t)N_NODES * 4);
  float* stats = (float*)carve(1024 * 4);
  float* cnt   = (float*)carve(NGRAPH * 4);

  auto cvt = [&](const float* s, bf16* d, int n) {
    f32_to_bf16<<<(n + 255) / 256, 256, 0, stream>>>(s, d, n);
  };
  auto wT = [&](const float* W, bf16* Wt, int K, int N) {   // B-side: transpose
    transpose_to_bf16<<<(K * N + 255) / 256, 256, 0, stream>>>(W, Wt, K, N);
  };
  auto gemm = [&](const bf16* A, const bf16* Bt, const float* bias, float* Cc,
                  int M, int Nn, int Kk) {
    gemm_bf16_wmma<<<dim3(M / 64, Nn / 64), 128, 0, stream>>>(A, Bt, bias, Cc, M, Nn, Kk);
  };
  auto bn = [&](const float* X, const float* g, const float* b, bf16* Y, int C) {
    zero_f32<<<(2 * C + 255) / 256, 256, 0, stream>>>(stats, 2 * C);
    colstats<<<dim3((C + 255) / 256, 64), 256, 0, stream>>>(X, stats, N_NODES, C);
    bn_relu_bf16<<<(N_NODES * C) / 256, 256, 0, stream>>>(X, stats, g, b, Y, N_NODES, C);
  };

  // graph ranges (batch is fixed for the whole net)
  ranges_kernel<<<N_NODES / 256, 256, 0, stream>>>(batch, gs, ge, N_NODES);

  // ---- node MLP: 64 -> 256 (BN+ReLU) -> 128
  cvt(x, hbf, N_NODES * 64);
  wT(nW1, WU, 64, 256);
  gemm(hbf, WU, nb1, bufA, N_NODES, 256, 64);
  bn(bufA, ng1, nbe1, hbf, 256);
  wT(nW2, WU, 256, 128);
  gemm(hbf, WU, nb2, bufA, N_NODES, 128, 256);
  cvt(bufA, hbf, N_NODES * 128);
  rownorm2<<<N_NODES / 256, 256, 0, stream>>>(hbf, d2, N_NODES, 128);

  // ---- three DynamicEdgeConv layers
  const int ci[3] = {128, 256, 512};
  const int co[3] = {256, 512, 512};
  for (int l = 0; l < 3; ++l) {
    knn_wmma<<<N_NODES / 16, 32, 0, stream>>>(hbf, d2, batch, gs, ge, idx, ci[l]);
    prep_conv_w<<<(ci[l] * co[l] + 255) / 256, 256, 0, stream>>>(cW[l], WU, WV, ci[l], co[l]);
    gemm(hbf, WU, cb[l], bufP, N_NODES, co[l], ci[l]);   // P = H*(Wt-Wb) + b
    gemm(hbf, WV, nullptr, bufQ, N_NODES, co[l], ci[l]); // Q = H*Wb
    zero_f32<<<(2 * co[l] + 255) / 256, 256, 0, stream>>>(stats, 2 * co[l]);
    msg_stats<<<dim3((co[l] + 255) / 256, 128), 256, 0, stream>>>(bufP, bufQ, idx, stats, co[l]);
    msg_max_relu<<<(N_NODES * co[l]) / 256, 256, 0, stream>>>(bufP, bufQ, idx, stats,
                                                              cg[l], cbe[l], bufA, co[l]);
    bn(bufA, cgn[l], cbn[l], hbf, co[l]);                // outer BN + ReLU -> next h (bf16)
    rownorm2<<<N_NODES / 256, 256, 0, stream>>>(hbf, d2, N_NODES, co[l]);
  }

  // ---- final MLP: 512 -> 512 (BN+ReLU) -> 256
  wT(mW1, WU, 512, 512);
  gemm(hbf, WU, mb1, bufA, N_NODES, 512, 512);
  bn(bufA, mg1, mbe1, hbf, 512);
  wT(mW2, WU, 512, 256);
  gemm(hbf, WU, mb2, bufB, N_NODES, 256, 512);

  // ---- global mean pool
  float* out = (float*)d_out;
  zero_f32<<<(NGRAPH * 256 + 255) / 256, 256, 0, stream>>>(out, NGRAPH * 256);
  zero_f32<<<1, 64, 0, stream>>>(cnt, NGRAPH);
  pool_accum<<<(N_NODES * 256) / 256, 256, 0, stream>>>(bufB, batch, out, cnt);
  pool_div<<<NGRAPH, 256, 0, stream>>>(out, cnt);
}
